// GraphFeatureEncoder_3959959847038
// MI455X (gfx1250) — matrix operand order repaired and checked by reading, not compiled, not run on hardware
//
#include <hip/hip_runtime.h>
#include <hip/hip_bf16.h>
#include <math.h>

typedef __attribute__((ext_vector_type(16))) __bf16 v16bf;
typedef __attribute__((ext_vector_type(8)))  float  v8f;
typedef __attribute__((ext_vector_type(4)))  __bf16 bf16x4;

constexpr int NN   = 20000;           // nodes
constexpr int NE   = 320000;          // edges (without self loops)
constexpr int ETOT = NE + NN;         // + self loops
constexpr int DK   = 128;             // feature dim (in == out == 128)
constexpr int NH   = 12;              // heads
constexpr int MOUT = NH * DK;         // 1536
constexpr float BN_EPS = 1e-5f;

// ---------------------------------------------------------------------------
// Per-node per-head linear: Y[N,1536] = X[N,128] * Wlin^T, bf16 WMMA, f32 acc.
// Block tile: 32 rows x 128 cols, K=128. 8 waves; wave -> (mt, 2 n-tiles).
// LDS holds A/B tiles pre-swizzled into the exact 16x16x32 bf16 fragment
// layout (ISA 7.12.2) so each lane reads its 16 elements contiguously.
// ---------------------------------------------------------------------------
__global__ __launch_bounds__(256)
void fsc_gemm_bf16(const float* __restrict__ X, const float* __restrict__ W,
                   __bf16* __restrict__ Y) {
  __shared__ v16bf aF[4][2][32];   // [kb][m-tile][lane]  : 8 KB
  __shared__ v16bf bF[4][8][32];   // [kb][n-tile][lane]  : 32 KB
  const int t    = threadIdx.x;
  const int lane = t & 31;
  const int wave = t >> 5;
  const int mBase = blockIdx.x * 32;
  const int cBase = blockIdx.y * 128;

  // Stage X tile (32x128), fp32 -> bf16, into A-fragment layout.
  // A 16-bit 16x32: lanes0-15 M=row; VGPR0..3 K=0..7, VGPR4..7 K=16..23;
  // lanes16-31 same M, K=8..15 / 24..31.
#pragma unroll
  for (int i = 0; i < 16; ++i) {
    int e = t + i * 256;                 // 0..4095
    int m = e >> 7, k = e & 127;
    float xv = X[(size_t)(mBase + m) * DK + k];
    int mt = m >> 4, mm = m & 15;
    int kb = k >> 5, kk = k & 31;
    int hi = (kk >> 3) & 1;
    int ln = mm + hi * 16;
    int fe = (kk & 7) + ((kk & 16) ? 8 : 0);
    ((__bf16*)&aF[kb][mt][ln])[fe] = (__bf16)xv;
  }
  // Stage W tile: B[kk][n] = Wlin[cBase+n][k]. B 32x16: lanes0-15 col=n
  // hold K=0..7 / 16..23, lanes16-31 hold K=8..15 / 24..31, 2 K per VGPR.
#pragma unroll
  for (int i = 0; i < 64; ++i) {
    int e = t + i * 256;                 // 0..16383
    int n = e >> 7, k = e & 127;
    float wv = W[(size_t)(cBase + n) * DK + k];
    int nt = n >> 4, nn = n & 15;
    int kb = k >> 5, kk = k & 31;
    int ln = ((kk >> 3) & 1) ? nn + 16 : nn;
    int fe = (kk & 7) + ((kk & 16) ? 8 : 0);
    ((__bf16*)&bF[kb][nt][ln])[fe] = (__bf16)wv;
  }
  __syncthreads();

  const int mt  = wave >> 2;
  const int nt0 = (wave & 3) * 2;
  v8f c0 = {}; v8f c1 = {};
#pragma unroll
  for (int kb = 0; kb < 4; ++kb) {
    v16bf a  = aF[kb][mt][lane];
    v16bf b0 = bF[kb][nt0][lane];
    v16bf b1 = bF[kb][nt0 + 1][lane];
    c0 = __builtin_amdgcn_wmma_f32_16x16x32_bf16(false, a, false, b0,
                                                 (short)0, c0, false, false);
    c1 = __builtin_amdgcn_wmma_f32_16x16x32_bf16(false, a, false, b1,
                                                 (short)0, c1, false, false);
  }
  // D layout: VGPR r, lanes0-15 -> M=r, lanes16-31 -> M=r+8; N = lane&15.
  const int row0 = mBase + mt * 16 + ((lane >> 4) << 3);
  const int col  = cBase + (lane & 15);
#pragma unroll
  for (int r = 0; r < 8; ++r) {
    size_t ro = (size_t)(row0 + r) * MOUT;
    Y[ro + col + nt0 * 16]       = (__bf16)c0[r];
    Y[ro + col + (nt0 + 1) * 16] = (__bf16)c1[r];
  }
}

// ---------------------------------------------------------------------------
// Per-node attention pre-projection: U[n,h] = sum_k X[n,k]*Wu[h,k]
// ---------------------------------------------------------------------------
__global__ __launch_bounds__(128)
void prep_kernel(const float* __restrict__ X, const float* __restrict__ Wu,
                 float* __restrict__ U) {
  __shared__ float xr[DK];
  const int n = blockIdx.x, t = threadIdx.x;
  xr[t] = X[(size_t)n * DK + t];
  __syncthreads();
  if (t < NH) {
    float s = 0.f;
#pragma unroll 8
    for (int k = 0; k < DK; ++k) s += xr[k] * Wu[t * DK + k];
    U[(size_t)n * NH + t] = s;
  }
}

// ---------------------------------------------------------------------------
// Degree (edge part; +1 self loop added in finalize)
// ---------------------------------------------------------------------------
__global__ void deg_kernel(const int* __restrict__ ei, float* __restrict__ deg) {
  int e = blockIdx.x * blockDim.x + threadIdx.x;
  if (e < NE) atomicAdd(&deg[ei[NE + e]], 1.0f);
}

// ---------------------------------------------------------------------------
// One wave per edge: 12-head softmax via wave32 shuffles, then
// agg[dst] += sum_h att_h * Y[src, h*128 .. h*128+127] (fp32 atomics).
// ---------------------------------------------------------------------------
__global__ __launch_bounds__(256)
void fsc_edge(const int* __restrict__ ei, const float* __restrict__ U,
              const float* __restrict__ cH, const __bf16* __restrict__ Y,
              float* __restrict__ agg) {
  const int lane = threadIdx.x & 31;
  const int wave = threadIdx.x >> 5;
  const int e = blockIdx.x * 8 + wave;
  if (e >= ETOT) return;
  int s, d;
  if (e < NE) { s = ei[e]; d = ei[NE + e]; } else { s = e - NE; d = s; }

  float t = -3.0e38f;
  if (lane < NH) t = U[(size_t)d * NH + lane] - U[(size_t)s * NH + lane] + cH[lane];
  float m = t;
  for (int off = 16; off > 0; off >>= 1) m = fmaxf(m, __shfl_xor(m, off));
  float ex  = (lane < NH) ? __expf(t - m) : 0.0f;
  float sum = ex;
  for (int off = 16; off > 0; off >>= 1) sum += __shfl_xor(sum, off);
  const float att = ex / sum;

  const bf16x4* yrow = (const bf16x4*)(Y + (size_t)s * MOUT);
  __builtin_prefetch(yrow, 0, 3);
  float a0 = 0.f, a1 = 0.f, a2 = 0.f, a3 = 0.f;
#pragma unroll
  for (int h = 0; h < NH; ++h) {
    float a = __shfl(att, h);
    bf16x4 v = yrow[h * 32 + lane];
    a0 += a * (float)v[0];
    a1 += a * (float)v[1];
    a2 += a * (float)v[2];
    a3 += a * (float)v[3];
  }
  float* dst = agg + (size_t)d * DK + lane * 4;
  atomicAdd(dst + 0, a0);
  atomicAdd(dst + 1, a1);
  atomicAdd(dst + 2, a2);
  atomicAdd(dst + 3, a3);
}

// ---------------------------------------------------------------------------
// out = agg / (deg+1) + b, optional ReLU
// ---------------------------------------------------------------------------
__global__ void finalize_kernel(const float* __restrict__ agg,
                                const float* __restrict__ deg,
                                const float* __restrict__ bias,
                                float* __restrict__ out, int relu) {
  int i = blockIdx.x * blockDim.x + threadIdx.x;
  if (i >= NN * DK) return;
  int n = i >> 7, ch = i & 127;
  float v = agg[i] / (deg[n] + 1.0f) + bias[ch];
  if (relu) v = fmaxf(v, 0.0f);
  out[i] = v;
}

// ---------------------------------------------------------------------------
// BatchNorm (training-mode biased stats): one block per channel
// ---------------------------------------------------------------------------
__global__ __launch_bounds__(256)
void bnstats_kernel(const float* __restrict__ X, float* __restrict__ mean,
                    float* __restrict__ rstd) {
  __shared__ float rs[256], rq[256];
  const int ch = blockIdx.x, t = threadIdx.x;
  float s = 0.f, q = 0.f;
  for (int n = t; n < NN; n += 256) {
    float v = X[(size_t)n * DK + ch];
    s += v; q += v * v;
  }
  rs[t] = s; rq[t] = q;
  __syncthreads();
  for (int off = 128; off > 0; off >>= 1) {
    if (t < off) { rs[t] += rs[t + off]; rq[t] += rq[t + off]; }
    __syncthreads();
  }
  if (t == 0) {
    float mm = rs[0] / (float)NN;
    float var = rq[0] / (float)NN - mm * mm;
    mean[ch] = mm;
    rstd[ch] = rsqrtf(var + BN_EPS);
  }
}

__global__ void bnapply_kernel(const float* __restrict__ X,
                               const float* __restrict__ mean,
                               const float* __restrict__ rstd,
                               const float* __restrict__ g,
                               const float* __restrict__ b,
                               float* __restrict__ out) {
  int i = blockIdx.x * blockDim.x + threadIdx.x;
  if (i >= NN * DK) return;
  int ch = i & 127;
  out[i] = (X[i] - mean[ch]) * rstd[ch] * g[ch] + b[ch];
}

// ---------------------------------------------------------------------------
extern "C" void kernel_launch(void* const* d_in, const int* in_sizes, int n_in,
                              void* d_out, int out_size, void* d_ws, size_t ws_size,
                              hipStream_t stream) {
  (void)in_sizes; (void)n_in; (void)out_size; (void)ws_size;
  const float* x  = (const float*)d_in[0];
  const int*   ei = (const int*)d_in[1];
  const float* Wl[3]  = { (const float*)d_in[2],  (const float*)d_in[6],  (const float*)d_in[10] };
  const float* Wu[3]  = { (const float*)d_in[3],  (const float*)d_in[7],  (const float*)d_in[11] };
  const float* cc[3]  = { (const float*)d_in[4],  (const float*)d_in[8],  (const float*)d_in[12] };
  const float* bb[3]  = { (const float*)d_in[5],  (const float*)d_in[9],  (const float*)d_in[13] };
  const float* bg[2]  = { (const float*)d_in[14], (const float*)d_in[16] };
  const float* bnb[2] = { (const float*)d_in[15], (const float*)d_in[17] };

  char* ws = (char*)d_ws;
  size_t off = 0;
  auto carve = [&](size_t bytes) -> void* {
    void* p = ws + off;
    off += (bytes + 255) & ~(size_t)255;
    return p;
  };
  float*  deg = (float*)carve((size_t)NN * 4);
  float*  U   = (float*)carve((size_t)NN * NH * 4);
  __bf16* Yb  = (__bf16*)carve((size_t)NN * MOUT * 2);
  float*  agg = (float*)carve((size_t)NN * DK * 4);
  float*  h1  = (float*)carve((size_t)NN * DK * 4);
  float*  h2  = (float*)carve((size_t)NN * DK * 4);
  float*  bnm = (float*)carve(DK * 4);
  float*  bnr = (float*)carve(DK * 4);

  hipMemsetAsync(deg, 0, (size_t)NN * 4, stream);
  deg_kernel<<<(NE + 255) / 256, 256, 0, stream>>>(ei, deg);

  const float* in = x;
  for (int li = 0; li < 3; ++li) {
    prep_kernel<<<NN, 128, 0, stream>>>(in, Wu[li], U);
    dim3 g(NN / 32, MOUT / 128);
    fsc_gemm_bf16<<<g, 256, 0, stream>>>(in, Wl[li], Yb);
    hipMemsetAsync(agg, 0, (size_t)NN * DK * 4, stream);
    fsc_edge<<<ETOT / 8, 256, 0, stream>>>(ei, U, cc[li], Yb, agg);
    float* outp = (li == 2) ? (float*)d_out : h1;
    finalize_kernel<<<(NN * DK + 255) / 256, 256, 0, stream>>>(
        agg, deg, bb[li], outp, (li < 2) ? 1 : 0);
    if (li < 2) {
      bnstats_kernel<<<DK, 256, 0, stream>>>(h1, bnm, bnr);
      bnapply_kernel<<<(NN * DK + 255) / 256, 256, 0, stream>>>(
          h1, bnm, bnr, bg[li], bnb[li], h2);
      in = h2;
    }
  }
}
